// ReflectiveCore_2465311228338
// MI455X (gfx1250) — matrix operand order repaired and checked by reading, not compiled, not run on hardware
//
#include <hip/hip_runtime.h>
#include <hip/hip_bf16.h>

#define B_ 4
#define T_ 1024
#define C_ 64
#define L_ (T_ + C_)      // 1088
#define D_ 768
#define H_ 12
#define DFF_ 3072
#define NL_ 4
#define V_ 1000
#define S_ 4
#define VS_ (V_ * S_)     // 4000
#define M_ (B_ * L_)      // 4352

typedef _Float16 h16;
typedef __attribute__((ext_vector_type(16))) _Float16 v16h;
typedef __attribute__((ext_vector_type(8)))  _Float16 v8h;
typedef __attribute__((ext_vector_type(8)))  float    v8f;
typedef __attribute__((ext_vector_type(4)))  int      v4i;

union V16 { v16h v; v8h h[2]; };

enum { MODE_H16 = 0, MODE_QKV = 1, MODE_RELU = 2, MODE_RES = 3, MODE_LOGITS = 4 };

// ---------------------------------------------------------------------------
// CDNA5 async global->LDS copy (ASYNCcnt path), with sync fallback
// ---------------------------------------------------------------------------
#if defined(__HIP_DEVICE_COMPILE__)
#if __has_builtin(__builtin_amdgcn_global_load_async_to_lds_b128)
#define ASYNC_MODE 1
#else
#define ASYNC_MODE 0
#endif
#else
#define ASYNC_MODE 0
#endif

__device__ __forceinline__ void async_cp16(h16* lds, const h16* g) {
#if ASYNC_MODE
    // signature (probe-confirmed arg1): (int4* global_src, int4* lds_dst, imm offset, imm cpol)
    __builtin_amdgcn_global_load_async_to_lds_b128((v4i*)g, (v4i*)lds, 0, 0);
#else
    *(v8h*)lds = *(const v8h*)g;
#endif
}

__device__ __forceinline__ void wait_async() {
#if ASYNC_MODE
#if __has_builtin(__builtin_amdgcn_s_wait_asynccnt)
    __builtin_amdgcn_s_wait_asynccnt(0);
#else
    asm volatile("s_wait_asynccnt 0" ::: "memory");
#endif
#endif
}

// ---------------------------------------------------------------------------
// f32 -> f16 weight conversion (grid-stride)
// ---------------------------------------------------------------------------
__global__ void f2h_kernel(const float* __restrict__ a, h16* __restrict__ o, size_t n) {
    size_t i = (size_t)blockIdx.x * blockDim.x + threadIdx.x;
    size_t stride = (size_t)gridDim.x * blockDim.x;
    for (; i < n; i += stride) o[i] = (h16)a[i];
}

// ---------------------------------------------------------------------------
// Embedding gather + prefix concat + positional add
// ---------------------------------------------------------------------------
__global__ __launch_bounds__(256)
void embed_kernel(const int* __restrict__ tok, const int* __restrict__ st,
                  const float* __restrict__ prefix, const float* __restrict__ in_w,
                  const float* __restrict__ in_b, const float* __restrict__ pos,
                  float* __restrict__ x32, h16* __restrict__ xh, h16* __restrict__ memh) {
    int m = blockIdx.x;                // 0..M_-1
    int b = m / L_, l = m % L_;
    int tid = threadIdx.x;
    int col = 0;
    if (l >= C_) col = tok[b * T_ + (l - C_)] * S_ + st[b];
#pragma unroll
    for (int i = 0; i < 3; i++) {
        int d = tid + i * 256;
        float v;
        if (l < C_) v = prefix[((size_t)b * C_ + l) * D_ + d];
        else        v = in_w[(size_t)d * VS_ + col] + in_b[d];
        v += pos[(size_t)l * D_ + d];
        x32[(size_t)m * D_ + d] = v;
        xh[(size_t)m * D_ + d] = (h16)v;
        memh[(size_t)m * D_ + d] = (h16)v;
    }
}

// ---------------------------------------------------------------------------
// Tiled WMMA GEMM: Y = act(A[M,K] * W[N,K]^T + bias [+ res])
// block = 256 threads (8 waves, 2x4), tile 128x128, K-tile 64,
// double-buffered LDS filled by async global->LDS DMA.
// ---------------------------------------------------------------------------
__global__ __launch_bounds__(256)
void gemm_wmma(const h16* __restrict__ A, const h16* __restrict__ W,
               const float* __restrict__ bias,
               int M, int N, int K, int mode, int n_off,
               const float* __restrict__ res,
               float* __restrict__ of32, h16* __restrict__ oh16,
               h16* __restrict__ qh, h16* __restrict__ kh, h16* __restrict__ vh) {
    __shared__ __align__(16) h16 Asb[2][128][72];   // 144B pitch, 16B-aligned chunks
    __shared__ __align__(16) h16 Bsb[2][128][72];

    const int tid = threadIdx.x;
    const int lane = tid & 31, wave = tid >> 5;
    const int wm = wave >> 2, wn = wave & 3;        // 2 (M) x 4 (N) wave grid
    const int cr = lane & 15, hl = lane >> 4;
    const int m0 = blockIdx.y * 128, n0 = blockIdx.x * 128;

    v8f acc[4][2];
#pragma unroll
    for (int i = 0; i < 4; i++)
#pragma unroll
        for (int j = 0; j < 2; j++)
#pragma unroll
            for (int e = 0; e < 8; e++) acc[i][j][e] = 0.f;

    // cooperative copy: thread -> (row, 64B half-row); rows clamped for W edge
    // (garbage rows only feed output columns that the store-guard discards)
    const int r2 = tid >> 1;                 // 0..127
    const int cb = (tid & 1) * 32;           // half offset 0 / 32 (64B)
    int nr = n0 + r2; if (nr >= N) nr = N - 1;
    const h16* gA = A + (size_t)(m0 + r2) * K + cb;
    const h16* gB = W + (size_t)nr * K + cb;

    auto issue = [&](int kb, int buf) {
#pragma unroll
        for (int i = 0; i < 4; i++) {
            async_cp16(&Asb[buf][r2][cb + i * 8], gA + kb + i * 8);
            async_cp16(&Bsb[buf][r2][cb + i * 8], gB + kb + i * 8);
        }
    };

    issue(0, 0);
    const int nk = K >> 6;                   // K-tiles of 64
    for (int t = 0; t < nk; t++) {
        wait_async();                        // own tile-t DMA complete
        __syncthreads();                     // everyone's tile-t visible
        if (t + 1 < nk) issue((t + 1) << 6, (t + 1) & 1);
        const int buf = t & 1;
#pragma unroll
        for (int kk = 0; kk < 2; kk++) {
            V16 af[4], bf[2];
#pragma unroll
            for (int mf = 0; mf < 4; mf++) {
                int r = wm * 64 + mf * 16 + cr;
                af[mf].h[0] = *(const v8h*)&Asb[buf][r][kk * 32 + hl * 8];
                af[mf].h[1] = *(const v8h*)&Asb[buf][r][kk * 32 + 16 + hl * 8];
            }
#pragma unroll
            for (int nf = 0; nf < 2; nf++) {
                int r = wn * 32 + nf * 16 + cr;
                bf[nf].h[0] = *(const v8h*)&Bsb[buf][r][kk * 32 + hl * 8];
                bf[nf].h[1] = *(const v8h*)&Bsb[buf][r][kk * 32 + 16 + hl * 8];
            }
#pragma unroll
            for (int mf = 0; mf < 4; mf++)
#pragma unroll
                for (int nf = 0; nf < 2; nf++)
                    acc[mf][nf] = __builtin_amdgcn_wmma_f32_16x16x32_f16(
                        false, af[mf].v, false, bf[nf].v, (short)0, acc[mf][nf], false, false);
        }
    }

    // epilogue: C element (vr,lane) -> row vr + 8*(lane>>4), col lane&15
#pragma unroll
    for (int mf = 0; mf < 4; mf++) {
#pragma unroll
        for (int nf = 0; nf < 2; nf++) {
#pragma unroll
            for (int vr = 0; vr < 8; vr++) {
                int m = m0 + wm * 64 + mf * 16 + hl * 8 + vr;
                int n = n0 + wn * 32 + nf * 16 + cr;
                if (n >= N || m >= M) continue;
                float val = acc[mf][nf][vr] + bias[n];
                if (mode == MODE_QKV) {
                    int b = m / L_, l = m % L_;
                    int ng = n + n_off;
                    if (ng < D_) {
                        int h = ng >> 6, d = ng & 63;
                        qh[(((size_t)b * H_ + h) * L_ + l) * 64 + d] = (h16)val;
                    } else if (ng < 2 * D_) {
                        int t2 = ng - D_; int h = t2 >> 6, d = t2 & 63;
                        kh[(((size_t)b * H_ + h) * L_ + l) * 64 + d] = (h16)val;
                    } else {
                        int t2 = ng - 2 * D_; int h = t2 >> 6, d = t2 & 63;
                        // V stored transposed: [B,H,64,L] so PV B-frags are contiguous
                        vh[(((size_t)b * H_ + h) * 64 + d) * L_ + l] = (h16)val;
                    }
                } else if (mode == MODE_RES) {
                    of32[(size_t)m * N + n] = val + res[(size_t)m * N + n];
                } else if (mode == MODE_RELU) {
                    oh16[(size_t)m * N + n] = (h16)fmaxf(val, 0.f);
                } else if (mode == MODE_LOGITS) {
                    int b = m / L_, l = m % L_;
                    if (l >= C_)
                        of32[((size_t)b * T_ + (l - C_)) * VS_ + n] = val;
                } else { // MODE_H16
                    oh16[(size_t)m * N + n] = (h16)val;
                }
            }
        }
    }
}

// ---------------------------------------------------------------------------
// Flash attention: block = 128 threads (4 waves), each wave owns 16 q rows.
// Q,K: [B,H,L,64] f16;  V: [B,H,64,L] f16 (transposed);  out: [B,L,D] f16
// ---------------------------------------------------------------------------
__global__ __launch_bounds__(128)
void attn_wmma(const h16* __restrict__ qh, const h16* __restrict__ kh,
               const h16* __restrict__ vh, h16* __restrict__ oh, int causal) {
    __shared__ __align__(16) h16 P[4][16][72];   // per-wave P tile, 144B pitch

    const int tid = threadIdx.x, lane = tid & 31, wave = tid >> 5;
    const int cr = lane & 15, hl = lane >> 4;
    const int bh = blockIdx.y;
    const int b = bh / H_, h = bh % H_;
    const int q0 = blockIdx.x * 64 + wave * 16;
    const size_t base  = (size_t)bh * L_ * 64;   // q/k base for this (b,h)
    const size_t vbase = (size_t)bh * 64 * L_;   // v^T base

    // Q fragments (A-matrix layout: row = lane&15, K split by lane>>4)
    V16 qa[2];
    const h16* qp = qh + base + (size_t)(q0 + cr) * 64;
#pragma unroll
    for (int kk = 0; kk < 2; kk++) {
        qa[kk].h[0] = *(const v8h*)(qp + kk * 32 + hl * 8);
        qa[kk].h[1] = *(const v8h*)(qp + kk * 32 + 16 + hl * 8);
    }

    v8f o[4], mI, lI;
#pragma unroll
    for (int i = 0; i < 4; i++)
#pragma unroll
        for (int e = 0; e < 8; e++) o[i][e] = 0.f;
#pragma unroll
    for (int e = 0; e < 8; e++) { mI[e] = -1e30f; lI[e] = 0.f; }

    const int nblk = L_ / 64;
    for (int j = 0; j < nblk; j++) {
        const int kb0 = j * 64;
        if (causal && kb0 > q0 + 15) break;     // fully masked for this wave

        // S = Q * K^T (scaled)
        v8f s[4];
#pragma unroll
        for (int nf = 0; nf < 4; nf++) {
#pragma unroll
            for (int e = 0; e < 8; e++) s[nf][e] = 0.f;
            const h16* kp = kh + base + (size_t)(kb0 + nf * 16 + cr) * 64;
#pragma unroll
            for (int kk = 0; kk < 2; kk++) {
                V16 bfr;
                bfr.h[0] = *(const v8h*)(kp + kk * 32 + hl * 8);
                bfr.h[1] = *(const v8h*)(kp + kk * 32 + 16 + hl * 8);
                s[nf] = __builtin_amdgcn_wmma_f32_16x16x32_f16(
                    false, qa[kk].v, false, bfr.v, (short)0, s[nf], false, false);
            }
        }
        // scale + analytic causal mask
#pragma unroll
        for (int nf = 0; nf < 4; nf++)
#pragma unroll
            for (int vr = 0; vr < 8; vr++) {
                float x = s[nf][vr] * 0.125f;     // 1/sqrt(64)
                if (causal) {
                    int qrow = q0 + hl * 8 + vr;
                    int kcol = kb0 + nf * 16 + cr;
                    if (kcol > qrow) x = -1e9f;
                }
                s[nf][vr] = x;
            }
        // online softmax: row reductions over 16 col-lanes per half-wave
        v8f mloc = s[0];
#pragma unroll
        for (int nf = 1; nf < 4; nf++)
#pragma unroll
            for (int vr = 0; vr < 8; vr++) mloc[vr] = fmaxf(mloc[vr], s[nf][vr]);
        v8f mN, alpha;
#pragma unroll
        for (int vr = 0; vr < 8; vr++) {
            float x = mloc[vr];
            for (int msk = 1; msk < 16; msk <<= 1) x = fmaxf(x, __shfl_xor(x, msk, 32));
            mN[vr] = fmaxf(mI[vr], x);
            alpha[vr] = __expf(mI[vr] - mN[vr]);
        }
        mI = mN;
#pragma unroll
        for (int nf = 0; nf < 4; nf++)
#pragma unroll
            for (int vr = 0; vr < 8; vr++) s[nf][vr] = __expf(s[nf][vr] - mN[vr]);
        v8f rs;
#pragma unroll
        for (int vr = 0; vr < 8; vr++) rs[vr] = s[0][vr] + s[1][vr] + s[2][vr] + s[3][vr];
#pragma unroll
        for (int vr = 0; vr < 8; vr++) {
            float x = rs[vr];
            for (int msk = 1; msk < 16; msk <<= 1) x += __shfl_xor(x, msk, 32);
            lI[vr] = lI[vr] * alpha[vr] + x;
        }
#pragma unroll
        for (int df = 0; df < 4; df++)
#pragma unroll
            for (int vr = 0; vr < 8; vr++) o[df][vr] *= alpha[vr];

        // C-layout -> A-fragment reshape via per-wave LDS tile
#pragma unroll
        for (int nf = 0; nf < 4; nf++)
#pragma unroll
            for (int vr = 0; vr < 8; vr++)
                P[wave][hl * 8 + vr][nf * 16 + cr] = (h16)s[nf][vr];

        // O += P * V  (V^T rows are contiguous along keys)
#pragma unroll
        for (int kk = 0; kk < 2; kk++) {
            V16 pfr;
            pfr.h[0] = *(const v8h*)&P[wave][cr][kk * 32 + hl * 8];
            pfr.h[1] = *(const v8h*)&P[wave][cr][kk * 32 + 16 + hl * 8];
#pragma unroll
            for (int df = 0; df < 4; df++) {
                const h16* vp = vh + vbase + (size_t)(df * 16 + cr) * L_ + kb0 + kk * 32 + hl * 8;
                V16 bfr;
                bfr.h[0] = *(const v8h*)(vp);
                bfr.h[1] = *(const v8h*)(vp + 16);
                o[df] = __builtin_amdgcn_wmma_f32_16x16x32_f16(
                    false, pfr.v, false, bfr.v, (short)0, o[df], false, false);
            }
        }
    }

    // write out (concat heads)
#pragma unroll
    for (int df = 0; df < 4; df++)
#pragma unroll
        for (int vr = 0; vr < 8; vr++) {
            float val = o[df][vr] / lI[vr];
            int q = q0 + hl * 8 + vr;
            oh[((size_t)b * L_ + q) * D_ + h * 64 + df * 16 + cr] = (h16)val;
        }
}

// ---------------------------------------------------------------------------
// LayerNorm: block (256) per row; writes fp32 master + f16 GEMM input
// ---------------------------------------------------------------------------
__global__ __launch_bounds__(256)
void ln_kernel(const float* __restrict__ t, const float* __restrict__ g,
               const float* __restrict__ bta, float* __restrict__ xo,
               h16* __restrict__ xh) {
    __shared__ float red[16];
    const int row = blockIdx.x, tid = threadIdx.x;
    const int wave = tid >> 5, lane = tid & 31;
    const float* p = t + (size_t)row * D_;
    float v[3], s = 0.f, ss = 0.f;
#pragma unroll
    for (int i = 0; i < 3; i++) {
        v[i] = p[tid + i * 256];
        s += v[i]; ss += v[i] * v[i];
    }
    for (int m = 1; m < 32; m <<= 1) { s += __shfl_xor(s, m, 32); ss += __shfl_xor(ss, m, 32); }
    if (lane == 0) { red[wave] = s; red[wave + 8] = ss; }
    __syncthreads();
    if (tid < 32) {
        float a = (lane < 8) ? red[lane] : 0.f;
        float b2 = (lane < 8) ? red[lane + 8] : 0.f;
        for (int m = 1; m < 8; m <<= 1) { a += __shfl_xor(a, m, 32); b2 += __shfl_xor(b2, m, 32); }
        if (lane == 0) { red[0] = a; red[1] = b2; }
    }
    __syncthreads();
    const float mean = red[0] * (1.f / D_);
    const float var = red[1] * (1.f / D_) - mean * mean;
    const float rstd = rsqrtf(var + 1e-5f);
#pragma unroll
    for (int i = 0; i < 3; i++) {
        int d = tid + i * 256;
        float y = (v[i] - mean) * rstd * g[d] + bta[d];
        xo[(size_t)row * D_ + d] = y;
        xh[(size_t)row * D_ + d] = (h16)y;
    }
}

// ---------------------------------------------------------------------------
extern "C" void kernel_launch(void* const* d_in, const int* in_sizes, int n_in,
                              void* d_out, int out_size, void* d_ws, size_t ws_size,
                              hipStream_t stream) {
    const int*   tok      = (const int*)d_in[0];
    const int*   st       = (const int*)d_in[1];
    const float* prefix   = (const float*)d_in[2];
    /* d_in[3] = mask: causal triu, applied analytically */
    const float* in_w     = (const float*)d_in[4];
    const float* in_b     = (const float*)d_in[5];
    const float* pos      = (const float*)d_in[6];
    const float* sa_qkv_w = (const float*)d_in[7];
    const float* sa_qkv_b = (const float*)d_in[8];
    const float* sa_out_w = (const float*)d_in[9];
    const float* sa_out_b = (const float*)d_in[10];
    const float* ca_qkv_w = (const float*)d_in[11];
    const float* ca_qkv_b = (const float*)d_in[12];
    const float* ca_out_w = (const float*)d_in[13];
    const float* ca_out_b = (const float*)d_in[14];
    const float* ln1_s    = (const float*)d_in[15];
    const float* ln1_b    = (const float*)d_in[16];
    const float* ln2_s    = (const float*)d_in[17];
    const float* ln2_b    = (const float*)d_in[18];
    const float* ln3_s    = (const float*)d_in[19];
    const float* ln3_b    = (const float*)d_in[20];
    const float* ff1_w    = (const float*)d_in[21];
    const float* ff1_b    = (const float*)d_in[22];
    const float* ff2_w    = (const float*)d_in[23];
    const float* ff2_b    = (const float*)d_in[24];
    const float* out_w    = (const float*)d_in[25];
    const float* out_b    = (const float*)d_in[26];
    (void)in_sizes; (void)n_in; (void)out_size; (void)ws_size;

    char* ws = (char*)d_ws;
    size_t off = 0;
    auto alloc = [&](size_t bytes) -> char* {
        char* p = ws + off;
        off += (bytes + 255) & ~(size_t)255;
        return p;
    };
    h16* sa_qkv_h = (h16*)alloc((size_t)NL_ * 3 * D_ * D_ * 2);
    h16* sa_out_h = (h16*)alloc((size_t)NL_ * D_ * D_ * 2);
    h16* ca_qkv_h = (h16*)alloc((size_t)NL_ * 3 * D_ * D_ * 2);
    h16* ca_out_h = (h16*)alloc((size_t)NL_ * D_ * D_ * 2);
    h16* ff1_h    = (h16*)alloc((size_t)NL_ * DFF_ * D_ * 2);
    h16* ff2_h    = (h16*)alloc((size_t)NL_ * D_ * DFF_ * 2);
    h16* outw_h   = (h16*)alloc((size_t)VS_ * D_ * 2);
    float* x32    = (float*)alloc((size_t)M_ * D_ * 4);
    float* tmp32  = (float*)alloc((size_t)M_ * D_ * 4);
    h16* xh       = (h16*)alloc((size_t)M_ * D_ * 2);
    h16* memh     = (h16*)alloc((size_t)M_ * D_ * 2);
    h16* attnh    = (h16*)alloc((size_t)M_ * D_ * 2);
    h16* ffh      = (h16*)alloc((size_t)M_ * DFF_ * 2);
    h16* qb       = (h16*)alloc((size_t)M_ * D_ * 2);
    h16* kb       = (h16*)alloc((size_t)M_ * D_ * 2);
    h16* vb       = (h16*)alloc((size_t)M_ * D_ * 2);

    auto cvt = [&](const float* src, h16* dst, size_t n) {
        f2h_kernel<<<2048, 256, 0, stream>>>(src, dst, n);
    };
    cvt(sa_qkv_w, sa_qkv_h, (size_t)NL_ * 3 * D_ * D_);
    cvt(sa_out_w, sa_out_h, (size_t)NL_ * D_ * D_);
    cvt(ca_qkv_w, ca_qkv_h, (size_t)NL_ * 3 * D_ * D_);
    cvt(ca_out_w, ca_out_h, (size_t)NL_ * D_ * D_);
    cvt(ff1_w,    ff1_h,    (size_t)NL_ * DFF_ * D_);
    cvt(ff2_w,    ff2_h,    (size_t)NL_ * D_ * DFF_);
    cvt(out_w,    outw_h,   (size_t)VS_ * D_);

    auto gemm = [&](const h16* A, const h16* W, const float* bias,
                    int M, int N, int K, int mode, int n_off,
                    const float* res, float* of32, h16* oh16) {
        dim3 g((N + 127) / 128, (M + 127) / 128);
        gemm_wmma<<<g, 256, 0, stream>>>(A, W, bias, M, N, K, mode, n_off,
                                         res, of32, oh16, qb, kb, vb);
    };

    embed_kernel<<<M_, 256, 0, stream>>>(tok, st, prefix, in_w, in_b, pos, x32, xh, memh);

    for (int i = 0; i < NL_; i++) {
        // ---- self attention ----
        gemm(xh, sa_qkv_h + (size_t)i * 3 * D_ * D_, sa_qkv_b + (size_t)i * 3 * D_,
             M_, 3 * D_, D_, MODE_QKV, 0, nullptr, nullptr, nullptr);
        attn_wmma<<<dim3(L_ / 64, B_ * H_), 128, 0, stream>>>(qb, kb, vb, attnh, 1);
        gemm(attnh, sa_out_h + (size_t)i * D_ * D_, sa_out_b + (size_t)i * D_,
             M_, D_, D_, MODE_RES, 0, x32, tmp32, nullptr);
        ln_kernel<<<M_, 256, 0, stream>>>(tmp32, ln1_s + i * D_, ln1_b + i * D_, x32, xh);

        // ---- cross attention (q from x, k/v from mem) ----
        gemm(xh, ca_qkv_h + (size_t)i * 3 * D_ * D_, ca_qkv_b + (size_t)i * 3 * D_,
             M_, D_, D_, MODE_QKV, 0, nullptr, nullptr, nullptr);
        gemm(memh, ca_qkv_h + (size_t)i * 3 * D_ * D_ + (size_t)D_ * D_,
             ca_qkv_b + (size_t)i * 3 * D_ + D_,
             M_, 2 * D_, D_, MODE_QKV, D_, nullptr, nullptr, nullptr);
        attn_wmma<<<dim3(L_ / 64, B_ * H_), 128, 0, stream>>>(qb, kb, vb, attnh, 0);
        gemm(attnh, ca_out_h + (size_t)i * D_ * D_, ca_out_b + (size_t)i * D_,
             M_, D_, D_, MODE_RES, 0, x32, tmp32, nullptr);
        ln_kernel<<<M_, 256, 0, stream>>>(tmp32, ln2_s + i * D_, ln2_b + i * D_, x32, xh);

        // ---- FFN ----
        gemm(xh, ff1_h + (size_t)i * DFF_ * D_, ff1_b + (size_t)i * DFF_,
             M_, DFF_, D_, MODE_RELU, 0, nullptr, nullptr, ffh);
        gemm(ffh, ff2_h + (size_t)i * D_ * DFF_, ff2_b + (size_t)i * D_,
             M_, D_, DFF_, MODE_RES, 0, x32, tmp32, nullptr);
        ln_kernel<<<M_, 256, 0, stream>>>(tmp32, ln3_s + i * D_, ln3_b + i * D_, x32, xh);
    }

    // ---- output head with [:, C:] slice ----
    gemm(xh, outw_h, out_b, M_, VS_, D_, MODE_LOGITS, 0, nullptr, (float*)d_out, nullptr);
}
